// LinearSelfAttention_72825465471230
// MI455X (gfx1250) — matrix operand order repaired and checked by reading, not compiled, not run on hardware
//
#include <hip/hip_runtime.h>
#include <hip/hip_bf16.h>
#include <stdint.h>

// ---------------------------------------------------------------------------
// LinearSelfAttention on MI455X (gfx1250): bf16 WMMA pipeline, f32 accum.
// LDS tiles staged in WMMA fragment order: each lane's operand is one
// 32-byte contiguous run (2x ds_load_b128), no per-element gathers.
// f32->bf16 via native converts (v_cvt_*_bf16_f32), RNE.
// ---------------------------------------------------------------------------

typedef __attribute__((ext_vector_type(16))) __bf16 v16bf;
typedef __attribute__((ext_vector_type(8)))  float  v8f;

#define B_   4
#define T_   4096
#define D_   1024
#define H_   16
#define HD_  64
#define M_   (B_ * T_)    // 16384 rows
#define N3_  (3 * D_)     // 3072 qkv cols
#define SCALE_ 0.125f     // HEAD_DIM^-0.5
#define LOG_ROPE_BASE 13.122363377404328f  // ln(500000)

__device__ __forceinline__ unsigned short f2bf(float f) {
  __bf16 h = (__bf16)f;                       // native cvt, round-nearest-even
  return __builtin_bit_cast(unsigned short, h);
}
__device__ __forceinline__ float bf2f(unsigned short h) {
  unsigned u = ((unsigned)h) << 16;
  return __builtin_bit_cast(float, u);
}
__device__ __forceinline__ unsigned pack2(float a, float b) {
  return (unsigned)f2bf(a) | ((unsigned)f2bf(b) << 16);  // -> v_cvt_pk_bf16_f32
}

__device__ __forceinline__ v8f wmma_bf16(v16bf a, v16bf b, v8f c) {
  return __builtin_amdgcn_wmma_f32_16x16x32_bf16(false, a, false, b,
                                                 (short)0, c, false, false);
}

// --- fragment-packed index maps (512 u16 per 16x32 A / 32x16 B subtile) ----
// A (MxK 16x32): lane = (m&15) + 16*((k>>3)&1), elem i = (k&7) + 8*(k>>4)
__device__ __forceinline__ int aidx(int m, int k) {
  return (((m & 15) + 16 * ((k >> 3) & 1)) << 4) + (k & 7) + 8 * (k >> 4);
}
// B (KxN 32x16): lane = (n&15) + 16*((k>>3)&1), elem i = (k&7) + 8*(k>>4)
__device__ __forceinline__ int bidx(int k, int n) {
  return (((n & 15) + 16 * ((k >> 3) & 1)) << 4) + (k & 7) + 8 * (k >> 4);
}

// One lane's whole fragment = 32 contiguous bytes at lane*16 u16.
__device__ __forceinline__ v16bf frag_ld(const unsigned short* p) {
  p += (threadIdx.x & 31) * 16;
  struct alignas(16) P { uint4 a, b; } t;
  t.a = *(const uint4*)p;
  t.b = *(const uint4*)(p + 8);
  return __builtin_bit_cast(v16bf, t);
}

// ---------------------------------------------------------------------------
// Kernel 1: qkv = x @ W_qkv fused with RoPE(q,k) + elu+1 feature maps.
// Block 256 thr (8 waves); block tile 128(M) x 64(N); wave tile 16x64.
// ---------------------------------------------------------------------------
__global__ __launch_bounds__(256) void qkv_rope_feat_kernel(
    const float* __restrict__ x, const float* __restrict__ Wqkv,
    unsigned short* __restrict__ qf, unsigned short* __restrict__ kf,
    unsigned short* __restrict__ vv) {
  __shared__ __align__(16) unsigned short lA[8 * 512];   // 8 A subtiles
  __shared__ __align__(16) unsigned short lB[4 * 512];   // 4 B subtiles
  const int tid = threadIdx.x;
  const int lane = tid & 31, wave = tid >> 5;
  const int rowBase = blockIdx.y * 128;
  const int colBase = blockIdx.x * 64;
  v8f acc[4] = {};

  const int arow = tid >> 1, ak = (tid & 1) * 16;        // A: 2 thr/row, 16 k each
  const int bn = (tid & 31) * 2, bk = (tid >> 5) * 4;    // B: 4k x 2n per thr

  for (int kk = 0; kk < D_; kk += 32) {
    // ---- stage A (x, f32 -> bf16, fragment-packed) ----
    const float* ap = x + (size_t)(rowBase + arow) * D_ + kk + ak;
    float4 f0 = ((const float4*)ap)[0], f1 = ((const float4*)ap)[1];
    float4 f2 = ((const float4*)ap)[2], f3 = ((const float4*)ap)[3];
    unsigned short* As = lA + (arow >> 4) * 512;
    uint4 r0 = {pack2(f0.x, f0.y), pack2(f0.z, f0.w), pack2(f1.x, f1.y), pack2(f1.z, f1.w)};
    uint4 r1 = {pack2(f2.x, f2.y), pack2(f2.z, f2.w), pack2(f3.x, f3.y), pack2(f3.z, f3.w)};
    *(uint4*)(As + aidx(arow, ak + 0)) = r0;   // k run [ak, ak+8)
    *(uint4*)(As + aidx(arow, ak + 8)) = r1;   // k run [ak+8, ak+16)
    // ---- stage B (W_qkv, fragment-packed) ----
    const float* bp = Wqkv + (size_t)(kk + bk) * N3_ + colBase + bn;
    float2 w0 = *(const float2*)bp;
    float2 w1 = *(const float2*)(bp + N3_);
    float2 w2 = *(const float2*)(bp + 2 * N3_);
    float2 w3 = *(const float2*)(bp + 3 * N3_);
    {
      uint2 c0 = {pack2(w0.x, w1.x), pack2(w2.x, w3.x)};
      uint2 c1 = {pack2(w0.y, w1.y), pack2(w2.y, w3.y)};
      *(uint2*)(lB + ((bn >> 4) * 512) + bidx(bk, bn)) = c0;
      *(uint2*)(lB + (((bn + 1) >> 4) * 512) + bidx(bk, bn + 1)) = c1;
    }
    if (kk + 32 < D_) {
      __builtin_prefetch(ap + 32, 0, 1);
      __builtin_prefetch(bp + (size_t)32 * N3_, 0, 1);
    }
    __syncthreads();
    v16bf a = frag_ld(lA + wave * 512);
#pragma unroll
    for (int s = 0; s < 4; ++s) {
      v16bf b = frag_ld(lB + s * 512);
      acc[s] = wmma_bf16(a, b, acc[s]);
    }
    __syncthreads();
  }

  // ---- epilogue: RoPE + feature map, scatter to (B,H,T,hd) bf16 ----
  const int seg = colBase >> 10;          // 0=q 1=k 2=v
  const int h   = (colBase & 1023) >> 6;
  const int n = lane & 15, half = lane >> 4;
  const float invf0 = __expf(-LOG_ROPE_BASE * (float)n        * (1.0f / 32.0f));
  const float invf1 = __expf(-LOG_ROPE_BASE * (float)(16 + n) * (1.0f / 32.0f));
  unsigned short* dstBase = (seg == 0) ? qf : (seg == 1) ? kf : vv;
#pragma unroll
  for (int r = 0; r < 8; ++r) {
    const int m = r + half * 8;
    const int rowG = rowBase + wave * 16 + m;
    const int b = rowG >> 12;
    const int t = rowG & (T_ - 1);
    float o0 = acc[0][r], o1 = acc[1][r], o2 = acc[2][r], o3 = acc[3][r];
    if (seg < 2) {
      const float a0 = (float)t * invf0, a1 = (float)t * invf1;
      const float c0 = __cosf(a0), s0 = __sinf(a0);
      const float c1 = __cosf(a1), s1 = __sinf(a1);
      float r0 = o0 * c0 - o2 * s0;
      float r1 = o1 * c1 - o3 * s1;
      float r2 = o2 * c0 + o0 * s0;
      float r3 = o3 * c1 + o1 * s1;
      const float sc = (seg == 0) ? SCALE_ : 1.0f;
      r0 *= sc; r1 *= sc; r2 *= sc; r3 *= sc;
      o0 = (r0 > 0.f) ? r0 + 1.f : __expf(r0);
      o1 = (r1 > 0.f) ? r1 + 1.f : __expf(r1);
      o2 = (r2 > 0.f) ? r2 + 1.f : __expf(r2);
      o3 = (r3 > 0.f) ? r3 + 1.f : __expf(r3);
    }
    const size_t dst = ((size_t)((b * H_ + h) * T_ + t)) * HD_;
    dstBase[dst + 0 * 16 + n] = f2bf(o0);
    dstBase[dst + 1 * 16 + n] = f2bf(o1);
    dstBase[dst + 2 * 16 + n] = f2bf(o2);
    dstBase[dst + 3 * 16 + n] = f2bf(o3);
  }
}

// ---------------------------------------------------------------------------
// Kernel 2: per (b,h): kv = k_feat^T @ v (64x64, K=T=4096) and k_sum.
// Block 512 thr (16 waves): wave (tm,tn) owns one 16x16 kv tile.
// K tile staged transposed into A-fragment order; k_sum accumulated from the
// staging registers and reduced with LDS float atomics (ds_add_f32).
// ---------------------------------------------------------------------------
__global__ __launch_bounds__(512) void kv_ksum_kernel(
    const unsigned short* __restrict__ kf, const unsigned short* __restrict__ vv,
    float* __restrict__ kv, float* __restrict__ ksum) {
  __shared__ __align__(16) unsigned short lK[4 * 512];   // A^T fragment-packed
  __shared__ __align__(16) unsigned short lV[4 * 512];   // B fragment-packed
  __shared__ float sSum[64];
  const int tid = threadIdx.x, lane = tid & 31, wave = tid >> 5;
  const int bh = blockIdx.x;
  const unsigned short* kp = kf + (size_t)bh * T_ * HD_;
  const unsigned short* vp = vv + (size_t)bh * T_ * HD_;
  if (tid < 64) sSum[tid] = 0.f;
  v8f acc = {};
  float kpart = 0.f;
  const int tm = wave >> 2, tn = wave & 3;
  const int kd = tid & 63, kq = (tid >> 6) * 4;   // lK: d, 4 consecutive t
  const int vn = (tid & 31) * 2, vk = (tid >> 5) * 2;  // lV: 2n x 2k

  for (int kk = 0; kk < T_; kk += 32) {
    // ---- stage K tile transposed: A[m=d, k=t] = kf[t, d] ----
    unsigned short e0 = kp[(size_t)(kk + kq + 0) * HD_ + kd];
    unsigned short e1 = kp[(size_t)(kk + kq + 1) * HD_ + kd];
    unsigned short e2 = kp[(size_t)(kk + kq + 2) * HD_ + kd];
    unsigned short e3 = kp[(size_t)(kk + kq + 3) * HD_ + kd];
    kpart += bf2f(e0) + bf2f(e1) + bf2f(e2) + bf2f(e3);
    uint2 kr = {(unsigned)e0 | ((unsigned)e1 << 16), (unsigned)e2 | ((unsigned)e3 << 16)};
    *(uint2*)(lK + (kd >> 4) * 512 + aidx(kd, kq)) = kr;
    // ---- stage V tile: B[k=t, n=d] ----
    ushort2 v0 = *(const ushort2*)(vp + (size_t)(kk + vk + 0) * HD_ + vn);
    ushort2 v1 = *(const ushort2*)(vp + (size_t)(kk + vk + 1) * HD_ + vn);
    *(unsigned*)(lV + (vn >> 4) * 512 + bidx(vk, vn)) =
        (unsigned)v0.x | ((unsigned)v1.x << 16);
    *(unsigned*)(lV + ((vn + 1) >> 4) * 512 + bidx(vk, vn + 1)) =
        (unsigned)v0.y | ((unsigned)v1.y << 16);
    if (kk + 32 < T_) __builtin_prefetch(kp + (size_t)(kk + 32 + kq) * HD_ + kd, 0, 1);
    __syncthreads();
    v16bf a = frag_ld(lK + tm * 512);
    v16bf b = frag_ld(lV + tn * 512);
    acc = wmma_bf16(a, b, acc);
    __syncthreads();
  }
  atomicAdd(&sSum[kd], kpart);          // ds_add_f32 reduction for k_sum
  const int n = tn * 16 + (lane & 15);
#pragma unroll
  for (int r = 0; r < 8; ++r) {
    const int m = tm * 16 + r + (lane >> 4) * 8;
    kv[(size_t)bh * HD_ * HD_ + m * HD_ + n] = acc[r];
  }
  __syncthreads();
  if (tid < 64) ksum[bh * HD_ + tid] = sSum[tid];
}

// ---------------------------------------------------------------------------
// Kernel 3: out = (q_feat @ kv) / max(q_feat.k_sum, 1e-6), bf16 (B,T,D) out.
// Block 256 thr (8 waves); 128-row slab of one (b,h); kv cached in LDS.
// ---------------------------------------------------------------------------
__global__ __launch_bounds__(256) void attn_out_kernel(
    const unsigned short* __restrict__ qf, const float* __restrict__ kv,
    const float* __restrict__ ksum, unsigned short* __restrict__ att) {
  __shared__ __align__(16) unsigned short lQ[8 * 1024];  // 8 waves x 2 ksteps
  __shared__ __align__(16) unsigned short lKV[2 * 2048]; // 2 ksteps x 4 subtiles
  __shared__ float sKs[64];
  __shared__ float sDen[128];
  const int tid = threadIdx.x, lane = tid & 31, wave = tid >> 5;
  const int bh = blockIdx.y;
  const int rowBase = blockIdx.x * 128;
  const unsigned short* qp = qf + (size_t)bh * T_ * HD_;

  {  // stage kv (f32 -> bf16, B-fragment-packed): 4k x 4n per thread
    const float* kvp = kv + (size_t)bh * HD_ * HD_;
    const int k4 = (tid >> 4) * 4, n4 = (tid & 15) * 4;
    float4 g0 = *(const float4*)(kvp + (k4 + 0) * HD_ + n4);
    float4 g1 = *(const float4*)(kvp + (k4 + 1) * HD_ + n4);
    float4 g2 = *(const float4*)(kvp + (k4 + 2) * HD_ + n4);
    float4 g3 = *(const float4*)(kvp + (k4 + 3) * HD_ + n4);
    unsigned short* base = lKV + (k4 >> 5) * 2048;
    const int kl = k4 & 31;
    uint2 c0 = {pack2(g0.x, g1.x), pack2(g2.x, g3.x)};
    uint2 c1 = {pack2(g0.y, g1.y), pack2(g2.y, g3.y)};
    uint2 c2 = {pack2(g0.z, g1.z), pack2(g2.z, g3.z)};
    uint2 c3 = {pack2(g0.w, g1.w), pack2(g2.w, g3.w)};
    *(uint2*)(base + ((n4 + 0) >> 4) * 512 + bidx(kl, n4 + 0)) = c0;
    *(uint2*)(base + ((n4 + 1) >> 4) * 512 + bidx(kl, n4 + 1)) = c1;
    *(uint2*)(base + ((n4 + 2) >> 4) * 512 + bidx(kl, n4 + 2)) = c2;
    *(uint2*)(base + ((n4 + 3) >> 4) * 512 + bidx(kl, n4 + 3)) = c3;
    if (tid < 64) sKs[tid] = ksum[bh * HD_ + tid];
  }
  {  // stage q slab 128x64 (A-fragment-packed per wave/kstep)
#pragma unroll
    for (int e = 0; e < 8; ++e) {
      const int idx = tid * 32 + e * 4;
      const int r2 = idx >> 6, c = idx & 63;
      const ushort4 u = *(const ushort4*)(qp + (size_t)(rowBase + r2) * HD_ + c);
      uint2 w = {(unsigned)u.x | ((unsigned)u.y << 16),
                 (unsigned)u.z | ((unsigned)u.w << 16)};
      *(uint2*)(lQ + (r2 >> 4) * 1024 + (c >> 5) * 512 + aidx(r2, c & 31)) = w;
    }
  }
  __syncthreads();
  if (tid < 128) {  // reciprocal denominators (reads packed layout)
    float dot = 0.f;
#pragma unroll
    for (int d = 0; d < 64; ++d)
      dot += bf2f(lQ[(tid >> 4) * 1024 + (d >> 5) * 512 + aidx(tid, d & 31)]) * sKs[d];
    sDen[tid] = 1.0f / fmaxf(dot, 1e-6f);
  }
  __syncthreads();

  v8f acc[4] = {};
#pragma unroll
  for (int ks = 0; ks < 2; ++ks) {
    v16bf a = frag_ld(lQ + wave * 1024 + ks * 512);
#pragma unroll
    for (int s = 0; s < 4; ++s) {
      v16bf b = frag_ld(lKV + ks * 2048 + s * 512);
      acc[s] = wmma_bf16(a, b, acc[s]);
    }
  }
  const int n = lane & 15, half = lane >> 4;
  const int b = bh >> 4, h = bh & 15;
#pragma unroll
  for (int r = 0; r < 8; ++r) {
    const int m = r + half * 8;
    const int t = rowBase + wave * 16 + m;
    const float dinv = sDen[wave * 16 + m];
    const size_t dst = ((size_t)(b * T_ + t)) * D_ + h * HD_;
#pragma unroll
    for (int s = 0; s < 4; ++s)
      att[dst + s * 16 + n] = f2bf(acc[s][r] * dinv);
  }
}

// ---------------------------------------------------------------------------
// Kernel 4: final = att @ W_out (16384x1024 x 1024x1024), f32 out.
// ---------------------------------------------------------------------------
__global__ __launch_bounds__(256) void out_proj_kernel(
    const unsigned short* __restrict__ att, const float* __restrict__ Wout,
    float* __restrict__ out) {
  __shared__ __align__(16) unsigned short lA[8 * 512];
  __shared__ __align__(16) unsigned short lB[4 * 512];
  const int tid = threadIdx.x, lane = tid & 31, wave = tid >> 5;
  const int rowBase = blockIdx.y * 128;
  const int colBase = blockIdx.x * 64;
  v8f acc[4] = {};
  const int arow = tid >> 1, ak = (tid & 1) * 16;
  const int bn = (tid & 31) * 2, bk = (tid >> 5) * 4;

  for (int kk = 0; kk < D_; kk += 32) {
    // A already bf16: 32B global load -> two b128 LDS stores (runs of 8 k)
    const unsigned short* ap = att + (size_t)(rowBase + arow) * D_ + kk + ak;
    uint4 q0 = ((const uint4*)ap)[0];   // k [ak, ak+8)
    uint4 q1 = ((const uint4*)ap)[1];   // k [ak+8, ak+16)
    unsigned short* As = lA + (arow >> 4) * 512;
    *(uint4*)(As + aidx(arow, ak + 0)) = q0;
    *(uint4*)(As + aidx(arow, ak + 8)) = q1;
    const float* bp = Wout + (size_t)(kk + bk) * D_ + colBase + bn;
    float2 w0 = *(const float2*)bp;
    float2 w1 = *(const float2*)(bp + D_);
    float2 w2 = *(const float2*)(bp + 2 * D_);
    float2 w3 = *(const float2*)(bp + 3 * D_);
    uint2 c0 = {pack2(w0.x, w1.x), pack2(w2.x, w3.x)};
    uint2 c1 = {pack2(w0.y, w1.y), pack2(w2.y, w3.y)};
    *(uint2*)(lB + ((bn >> 4) * 512) + bidx(bk, bn)) = c0;
    *(uint2*)(lB + (((bn + 1) >> 4) * 512) + bidx(bk, bn + 1)) = c1;
    if (kk + 32 < D_) {
      __builtin_prefetch(ap + 32, 0, 1);
      __builtin_prefetch(bp + (size_t)32 * D_, 0, 1);
    }
    __syncthreads();
    v16bf a = frag_ld(lA + wave * 512);
#pragma unroll
    for (int s = 0; s < 4; ++s) {
      v16bf b = frag_ld(lB + s * 512);
      acc[s] = wmma_bf16(a, b, acc[s]);
    }
    __syncthreads();
  }
  const int n = lane & 15, half = lane >> 4;
#pragma unroll
  for (int r = 0; r < 8; ++r) {
    const int m = r + half * 8;
    const size_t dst = (size_t)(rowBase + wave * 16 + m) * D_ + colBase;
#pragma unroll
    for (int s = 0; s < 4; ++s) out[dst + s * 16 + n] = acc[s][r];
  }
}

// ---------------------------------------------------------------------------
extern "C" void kernel_launch(void* const* d_in, const int* in_sizes, int n_in,
                              void* d_out, int out_size, void* d_ws, size_t ws_size,
                              hipStream_t stream) {
  const float* x    = (const float*)d_in[0];   // (B,T,D)
  const float* Wqkv = (const float*)d_in[1];   // (D,3D)
  const float* Wout = (const float*)d_in[2];   // (D,D)
  float* out = (float*)d_out;                  // (B,T,D) f32

  char* ws = (char*)d_ws;
  size_t off = 0;
  auto carve = [&](size_t bytes) -> void* {
    void* p = (void*)(ws + off);
    off += (bytes + 255) & ~(size_t)255;
    return p;
  };
  const size_t featElems = (size_t)B_ * H_ * T_ * HD_;
  unsigned short* qf   = (unsigned short*)carve(featElems * 2);
  unsigned short* kf   = (unsigned short*)carve(featElems * 2);
  unsigned short* vv   = (unsigned short*)carve(featElems * 2);
  unsigned short* att  = (unsigned short*)carve((size_t)M_ * D_ * 2);
  float* kv   = (float*)carve((size_t)B_ * H_ * HD_ * HD_ * 4);
  float* ksum = (float*)carve((size_t)B_ * H_ * HD_ * 4);

  qkv_rope_feat_kernel<<<dim3(N3_ / 64, M_ / 128), 256, 0, stream>>>(x, Wqkv, qf, kf, vv);
  kv_ksum_kernel<<<dim3(B_ * H_), 512, 0, stream>>>(kf, vv, kv, ksum);
  attn_out_kernel<<<dim3(T_ / 128, B_ * H_), 256, 0, stream>>>(qf, kv, ksum, att);
  out_proj_kernel<<<dim3(D_ / 64, M_ / 128), 256, 0, stream>>>(att, Wout, out);
}
